// cait_models_48799418417807
// MI455X (gfx1250) — matrix operand order repaired and checked by reading, not compiled, not run on hardware
//
#include <hip/hip_runtime.h>
#include <hip/hip_bf16.h>
#include <math.h>

typedef __bf16 bf16;
typedef __attribute__((ext_vector_type(16))) __bf16 v16bf;
typedef __attribute__((ext_vector_type(8)))  __bf16 v8bf;
typedef __attribute__((ext_vector_type(8)))  float  v8f;

#define DEV __device__ __forceinline__

static constexpr int Dm   = 384;
static constexpr int Hn   = 8;
static constexpr int HD   = 48;
static constexpr int NP   = 196;
static constexpr int NPAD = 224;          // padded key dim (multiple of 32)
static constexpr int PDm  = 768;
static constexpr int MLPm = 1536;
static constexpr int NCLS = 1000;
static constexpr int Bb   = 32;
static constexpr int ROWS  = Bb * NP;     // 6272
static constexpr int UROWS = Bb * (NP+1); // 6304
static constexpr float SCALE = 0.14433756729740643f; // 48^-0.5

// ---------------- input index map (setup_inputs insertion order) ----------
enum {
  I_X = 0, I_PATCH_W, I_PATCH_B, I_CLS, I_POS,
  I_N1W, I_N1B, I_QKVW, I_PROJW, I_PROJB, I_PLW, I_PLB, I_PWW, I_PWB,
  I_N2W, I_N2B, I_FC1W, I_FC1B, I_FC2W, I_FC2B, I_G1, I_G2,
  I_CN1W, I_CN1B, I_CQW, I_CKW, I_CVW, I_CPROJW, I_CPROJB, I_CN2W, I_CN2B,
  I_CFC1W, I_CFC1B, I_CFC2W, I_CFC2B, I_CG1, I_CG2,
  I_NORMW, I_NORMB, I_HEADW, I_HEADB
};

// ================= fast (fully-aligned) tile loaders ======================
// A tile 16x32 (MxK), wave32 layout: lanes 0-15 row=lane: K[0..7],K[16..23];
// lanes 16-31 row=lane-16: K[8..15],K[24..31]. Two b128 loads, no branches.
DEV v16bf load_a_fast(const bf16* __restrict__ A, long lda, int row0, int k0) {
  int lane = threadIdx.x & 31;
  const bf16* base = A + (long)(row0 + (lane & 15)) * lda + k0 + (lane >> 4) * 8;
  v8bf lo = *(const v8bf*)base;
  v8bf hi = *(const v8bf*)(base + 16);
  return __builtin_shufflevector(lo, hi, 0,1,2,3,4,5,6,7,8,9,10,11,12,13,14,15);
}
// B tile 32x16 (KxN) from W stored [N,K]: lanes 0-15 col=lane: K[0..7],K[16..23];
// lanes 16-31 col=lane-16: K[8..15],K[24..31].
DEV v16bf load_b_fast(const bf16* __restrict__ W, long ldw, int col0, int k0) {
  int lane = threadIdx.x & 31;
  const bf16* base = W + (long)(col0 + (lane & 15)) * ldw + k0 + (lane >> 4) * 8;
  v8bf lo = *(const v8bf*)base;
  v8bf hi = *(const v8bf*)(base + 16);
  return __builtin_shufflevector(lo, hi, 0,1,2,3,4,5,6,7,8,9,10,11,12,13,14,15);
}

// ================= guarded tile loaders (ragged shapes) ===================
DEV void load_chunk8(const bf16* __restrict__ p, int nvalid, v16bf& dst, int off) {
  if (nvalid >= 8) {
    v8bf v = *(const v8bf*)p;
#pragma unroll
    for (int j = 0; j < 8; ++j) dst[off + j] = v[j];
  } else {
#pragma unroll
    for (int j = 0; j < 8; ++j) if (j < nvalid) dst[off + j] = p[j];
  }
}

DEV v16bf load_a_tile(const bf16* __restrict__ A, long lda,
                      int row0, int Mmax, int k0, int Kmax) {
  int lane = threadIdx.x & 31;
  int half = lane >> 4;
  int row  = row0 + (lane & 15);
  v16bf a = {};
  if (row < Mmax) {
    const bf16* base = A + (long)row * lda;
    int ka = k0 + half * 8;
    int kb = k0 + 16 + half * 8;
    load_chunk8(base + ka, Kmax - ka, a, 0);
    load_chunk8(base + kb, Kmax - kb, a, 8);
  }
  return a;
}

DEV v16bf load_bt_tile(const bf16* __restrict__ W, long ldw,
                       int col0, int Nmax, int k0, int Kmax) {
  int lane = threadIdx.x & 31;
  int half = lane >> 4;
  int n    = col0 + (lane & 15);
  v16bf b = {};
  if (n < Nmax) {
    int kb = k0 + half * 8;
    const bf16* base = W + (long)n * ldw + kb;
    load_chunk8(base,      Kmax - kb,        b, 0);
    load_chunk8(base + 16, Kmax - (kb + 16), b, 8);
  }
  return b;
}

// ================= aligned fast GEMM (bulk of the FLOPs) ==================
// Requires M%32==0 (per-wave 32 rows), N%64==0, K%32==0, no zero padding.
// Wave: 32x64 tile -> 8 accumulators, 8 WMMAs per K-step with B reuse.
template <bool GELU>
__global__ __launch_bounds__(256)
void k_gemm_al(const bf16* __restrict__ A, const bf16* __restrict__ Bm,
               const float* __restrict__ bias, const float* __restrict__ res,
               const float* __restrict__ gamma,
               float* __restrict__ outF, bf16* __restrict__ outB,
               int M, int N, int K, long lda, long ldb, long ldc) {
  int wave = threadIdx.x >> 5;
  int row0 = blockIdx.x * 64  + (wave & 1) * 32;
  int col0 = blockIdx.y * 256 + (wave >> 1) * 64;
  if (row0 >= M || col0 >= N) return;   // wave-uniform

  v8f acc[2][4] = {};
#pragma unroll 2
  for (int k0 = 0; k0 < K; k0 += 32) {
    v16bf a0 = load_a_fast(A, lda, row0,      k0);
    v16bf a1 = load_a_fast(A, lda, row0 + 16, k0);
#pragma unroll
    for (int t = 0; t < 4; ++t) {
      v16bf b = load_b_fast(Bm, ldb, col0 + 16 * t, k0);
      acc[0][t] = __builtin_amdgcn_wmma_f32_16x16x32_bf16(
          false, a0, false, b, (short)0, acc[0][t], false, false);
      acc[1][t] = __builtin_amdgcn_wmma_f32_16x16x32_bf16(
          false, a1, false, b, (short)0, acc[1][t], false, false);
    }
  }

  int lane = threadIdx.x & 31;
  int ccol = lane & 15;
#pragma unroll
  for (int i = 0; i < 2; ++i) {
    int rbase = row0 + i * 16 + (lane >> 4) * 8;
#pragma unroll
    for (int t = 0; t < 4; ++t) {
      int c = col0 + 16 * t + ccol;
      float bv = bias  ? bias[c]  : 0.f;
      float gv = gamma ? gamma[c] : 0.f;
#pragma unroll
      for (int r = 0; r < 8; ++r) {
        int rw = rbase + r;
        float v = acc[i][t][r] + bv;
        if constexpr (GELU) v = 0.5f * v * (1.f + erff(v * 0.70710678118654752f));
        long idx = (long)rw * ldc + c;
        if (res) v = res[idx] + gv * v;
        if (outF) outF[idx] = v;
        if (outB) outB[idx] = (bf16)v;
      }
    }
  }
}

// ================= guarded batched GEMM (ragged attention/head) ===========
// B always W[N,K] K-contiguous. grid.z batches: z -> (bb=z/gDiv, gg=z%gDiv).
template <bool GELU>
__global__ __launch_bounds__(256)
void k_gemm_g(const bf16* __restrict__ A, const bf16* __restrict__ Bm,
              const float* __restrict__ bias, const float* __restrict__ res,
              const float* __restrict__ gamma,
              float* __restrict__ outF, bf16* __restrict__ outB,
              int M, int N, int K, int KA, int KB,
              long lda, long ldb, long ldc, int gDiv,
              long sAb, long sAg, long sBb, long sBg, long sCb, long sCg) {
  int z  = blockIdx.z;
  int bb = z / gDiv, gg = z % gDiv;
  A  += (long)bb * sAb + (long)gg * sAg;
  Bm += (long)bb * sBb + (long)gg * sBg;
  long co = (long)bb * sCb + (long)gg * sCg;

  int wave = threadIdx.x >> 5;
  int row0 = blockIdx.x * 32  + (wave & 1) * 16;
  int col0 = blockIdx.y * 256 + (wave >> 1) * 64;
  if (row0 >= M || col0 >= N) return;   // wave-uniform exit

  v8f acc[4] = {};
  for (int k0 = 0; k0 < K; k0 += 32) {
    v16bf a = load_a_tile(A, lda, row0, M, k0, KA);
#pragma unroll
    for (int t = 0; t < 4; ++t) {
      v16bf bt = load_bt_tile(Bm, ldb, col0 + 16 * t, N, k0, KB);
      acc[t] = __builtin_amdgcn_wmma_f32_16x16x32_bf16(
          false, a, false, bt, (short)0, acc[t], false, false);
    }
  }

  int lane  = threadIdx.x & 31;
  int ccol  = lane & 15;
  int rbase = row0 + (lane >> 4) * 8;
#pragma unroll
  for (int t = 0; t < 4; ++t) {
    int c = col0 + 16 * t + ccol;
    if (c >= N) continue;
    float bv = bias  ? bias[c]  : 0.f;
    float gv = gamma ? gamma[c] : 0.f;
#pragma unroll
    for (int r = 0; r < 8; ++r) {
      int rw = rbase + r;
      if (rw >= M) continue;
      float v = acc[t][r] + bv;
      if constexpr (GELU) v = 0.5f * v * (1.f + erff(v * 0.70710678118654752f));
      if (res) v = res[(long)rw * ldc + c] + gv * v;
      long idx = co + (long)rw * ldc + c;
      if (outF) outF[idx] = v;
      if (outB) outB[idx] = (bf16)v;
    }
  }
}

// ---------------- support kernels -----------------------------------------
__global__ void k_cvt(const float* __restrict__ s, bf16* __restrict__ d, long n) {
  for (long i = blockIdx.x * (long)blockDim.x + threadIdx.x; i < n;
       i += (long)gridDim.x * blockDim.x)
    d[i] = (bf16)s[i];
}

// b c (h p1) (w p2) -> b (h w) (p1 p2 c), output bf16 [B*196, 768]
__global__ void k_patchify(const float* __restrict__ x, bf16* __restrict__ out, long n) {
  for (long i = blockIdx.x * (long)blockDim.x + threadIdx.x; i < n;
       i += (long)gridDim.x * blockDim.x) {
    int pd = (int)(i % PDm);
    long t = i / PDm;
    int np = (int)(t % NP);
    int b  = (int)(t / NP);
    int c  = pd % 3;
    int p2 = (pd / 3) & 15;
    int p1 = pd / 48;
    int ph = np / 14, pw = np % 14;
    long src = (((long)b * 3 + c) * 224 + (ph * 16 + p1)) * 224 + (pw * 16 + p2);
    out[i] = (bf16)x[src];
  }
}

__global__ void k_addpos(float* __restrict__ x, const float* __restrict__ pos, long n) {
  for (long i = blockIdx.x * (long)blockDim.x + threadIdx.x; i < n;
       i += (long)gridDim.x * blockDim.x)
    x[i] += pos[i % ((long)NP * Dm)];
}

// V slice of QKV -> vT[b][h][d][m] bf16, ld=NPAD, zero-padded m in [196,224)
__global__ void k_transpose_v(const bf16* __restrict__ qkv, bf16* __restrict__ vT, long n) {
  for (long i = blockIdx.x * (long)blockDim.x + threadIdx.x; i < n;
       i += (long)gridDim.x * blockDim.x) {
    int m  = (int)(i % NPAD);
    long t = i / NPAD;
    int d  = (int)(t % HD);
    long bh = t / HD;              // b*8 + h
    int h = (int)(bh & 7);
    int b = (int)(bh >> 3);
    bf16 v = (bf16)0.f;
    if (m < NP)
      v = qkv[((long)b * NP + m) * 1152 + 2 * Dm + h * HD + d];
    vT[i] = v;
  }
}

__global__ __launch_bounds__(128)
void k_layernorm(const float* __restrict__ x, const float* __restrict__ w,
                 const float* __restrict__ b, bf16* __restrict__ y, int D) {
  __shared__ float red[128];
  long row = blockIdx.x;
  const float* xr = x + row * D;
  int tid = threadIdx.x;
  float s = 0.f;
  for (int i = tid; i < D; i += 128) s += xr[i];
  red[tid] = s; __syncthreads();
  for (int o = 64; o > 0; o >>= 1) { if (tid < o) red[tid] += red[tid + o]; __syncthreads(); }
  float mean = red[0] / D; __syncthreads();
  float vs = 0.f;
  for (int i = tid; i < D; i += 128) { float d0 = xr[i] - mean; vs += d0 * d0; }
  red[tid] = vs; __syncthreads();
  for (int o = 64; o > 0; o >>= 1) { if (tid < o) red[tid] += red[tid + o]; __syncthreads(); }
  float rstd = rsqrtf(red[0] / D + 1e-5f); __syncthreads();
  for (int i = tid; i < D; i += 128)
    y[row * D + i] = (bf16)((xr[i] - mean) * rstd * w[i] + b[i]);
}

// talking-heads: pre-mix(+SCALE) -> softmax(8 heads) -> post-mix -> bf16 (padded)
__global__ __launch_bounds__(256)
void k_talking_softmax(const float* __restrict__ a1, bf16* __restrict__ a2,
                       const float* __restrict__ plw, const float* __restrict__ plb,
                       const float* __restrict__ pww, const float* __restrict__ pwb) {
  __shared__ float red[256];
  int bn = blockIdx.x;
  int b = bn / NP, n = bn % NP;
  int m = threadIdx.x;
  long hs = (long)NP * NPAD;
  const float* base = a1 + (long)b * Hn * hs + (long)n * NPAD;
  float av[8];
#pragma unroll
  for (int h = 0; h < 8; ++h) av[h] = (m < NP) ? base[h * hs + m] : 0.f;
  float pre[8];
#pragma unroll
  for (int g = 0; g < 8; ++g) {
    float s = 0.f;
#pragma unroll
    for (int h = 0; h < 8; ++h) s += av[h] * plw[g * 8 + h];
    pre[g] = s * SCALE + plb[g];
  }
  float soft[8];
  for (int g = 0; g < 8; ++g) {
    red[m] = (m < NP) ? pre[g] : -3.0e38f; __syncthreads();
    for (int o = 128; o > 0; o >>= 1) { if (m < o) red[m] = fmaxf(red[m], red[m + o]); __syncthreads(); }
    float mx = red[0]; __syncthreads();
    float e = (m < NP) ? expf(pre[g] - mx) : 0.f;
    red[m] = e; __syncthreads();
    for (int o = 128; o > 0; o >>= 1) { if (m < o) red[m] += red[m + o]; __syncthreads(); }
    float sm = red[0]; __syncthreads();
    soft[g] = e / sm;
  }
  if (m < NPAD) {
    long obase = (long)b * Hn * hs + (long)n * NPAD + m;
#pragma unroll
    for (int g = 0; g < 8; ++g) {
      float s = pwb[g];
#pragma unroll
      for (int h = 0; h < 8; ++h) s += soft[h] * pww[g * 8 + h];
      a2[(long)g * hs + obase] = (m < NP) ? (bf16)s : (bf16)0.f;
    }
  }
}

__global__ void k_init_cls(float* __restrict__ cls, const float* __restrict__ tok, long n) {
  for (long i = blockIdx.x * (long)blockDim.x + threadIdx.x; i < n;
       i += (long)gridDim.x * blockDim.x)
    cls[i] = tok[i % Dm];
}

__global__ void k_build_u(float* __restrict__ u, const float* __restrict__ cls,
                          const float* __restrict__ x, long n) {
  for (long i = blockIdx.x * (long)blockDim.x + threadIdx.x; i < n;
       i += (long)gridDim.x * blockDim.x) {
    int d = (int)(i % Dm);
    long r = i / Dm;
    int b = (int)(r / (NP + 1));
    int p = (int)(r % (NP + 1));
    u[i] = (p == 0) ? cls[(long)b * Dm + d]
                    : x[((long)b * NP + p - 1) * Dm + d];
  }
}

__global__ void k_gather_rows(bf16* __restrict__ qy, const bf16* __restrict__ y, long n) {
  for (long i = blockIdx.x * (long)blockDim.x + threadIdx.x; i < n;
       i += (long)gridDim.x * blockDim.x) {
    int b = (int)(i / Dm);
    qy[i] = y[(long)b * (NP + 1) * Dm + (i % Dm)];
  }
}

// class-attention: 1 query row per (b,h); softmax over 197 keys; out bf16
__global__ __launch_bounds__(256)
void k_ca_attn(const float* __restrict__ qc, const float* __restrict__ kc,
               const float* __restrict__ vc, bf16* __restrict__ o) {
  __shared__ float sc[256];
  __shared__ float red[256];
  int bh = blockIdx.x;
  int b = bh >> 3, h = bh & 7;
  int j = threadIdx.x;
  const float* q = qc + (long)b * Dm + h * HD;
  const float* kb = kc + (long)b * (NP + 1) * Dm + h * HD;
  float s = -3.0e38f;
  if (j < NP + 1) {
    float acc = 0.f;
#pragma unroll
    for (int d = 0; d < HD; ++d) acc += q[d] * kb[(long)j * Dm + d];
    s = acc * SCALE;
  }
  red[j] = s; __syncthreads();
  for (int o2 = 128; o2 > 0; o2 >>= 1) { if (j < o2) red[j] = fmaxf(red[j], red[j + o2]); __syncthreads(); }
  float mx = red[0]; __syncthreads();
  float e = (j < NP + 1) ? expf(s - mx) : 0.f;
  red[j] = e; __syncthreads();
  for (int o2 = 128; o2 > 0; o2 >>= 1) { if (j < o2) red[j] += red[j + o2]; __syncthreads(); }
  float sm = red[0]; __syncthreads();
  sc[j] = e / sm; __syncthreads();
  if (j < HD) {
    const float* vb = vc + (long)b * (NP + 1) * Dm + h * HD + j;
    float acc = 0.f;
    for (int t = 0; t < NP + 1; ++t) acc += sc[t] * vb[(long)t * Dm];
    o[(long)b * Dm + h * HD + j] = (bf16)acc;
  }
}

// ---------------- host-side GEMM dispatcher -------------------------------
static void launch_gemm(hipStream_t st, bool GELU,
                        const bf16* A, const bf16* Bm, const float* bias,
                        const float* res, const float* gamma,
                        float* outF, bf16* outB,
                        int M, int N, int K, int KA, int KB,
                        long lda, long ldb, long ldc,
                        int Z = 1, int gDiv = 1,
                        long sAb = 0, long sAg = 0, long sBb = 0, long sBg = 0,
                        long sCb = 0, long sCg = 0) {
  bool aligned = Z == 1 && (M % 32 == 0) && (N % 64 == 0) &&
                 (K % 32 == 0) && KA == K && KB == K;
  if (aligned) {
    dim3 grid((M + 63) / 64, (N + 255) / 256), blk(256);
    if (GELU)
      k_gemm_al<true><<<grid, blk, 0, st>>>(A, Bm, bias, res, gamma, outF, outB,
                                            M, N, K, lda, ldb, ldc);
    else
      k_gemm_al<false><<<grid, blk, 0, st>>>(A, Bm, bias, res, gamma, outF, outB,
                                             M, N, K, lda, ldb, ldc);
    return;
  }
  dim3 grid((M + 31) / 32, (N + 255) / 256, Z), blk(256);
  if (GELU)
    k_gemm_g<true><<<grid, blk, 0, st>>>(A, Bm, bias, res, gamma, outF, outB,
        M, N, K, KA, KB, lda, ldb, ldc, gDiv, sAb, sAg, sBb, sBg, sCb, sCg);
  else
    k_gemm_g<false><<<grid, blk, 0, st>>>(A, Bm, bias, res, gamma, outF, outB,
        M, N, K, KA, KB, lda, ldb, ldc, gDiv, sAb, sAg, sBb, sBg, sCb, sCg);
}

static inline int gsz(long n) { return (int)((n + 255) / 256); }

// ---------------- driver --------------------------------------------------
extern "C" void kernel_launch(void* const* d_in, const int* in_sizes, int n_in,
                              void* d_out, int out_size, void* d_ws, size_t ws_size,
                              hipStream_t stream) {
  (void)in_sizes; (void)n_in; (void)out_size; (void)ws_size;
  const float* IN[41];
  for (int i = 0; i < 41; ++i) IN[i] = (const float*)d_in[i];

  // bump allocator over workspace
  char* wsb = (char*)d_ws;
  size_t off = 0;
  auto alloc = [&](size_t bytes) -> void* {
    void* p = wsb + off;
    off = (off + bytes + 255) & ~(size_t)255;
    return p;
  };

  bf16* wPatch = (bf16*)alloc(384L * 768 * 2);
  bf16* wQkv   = (bf16*)alloc(12L * 1152 * 384 * 2);
  bf16* wProj  = (bf16*)alloc(12L * 384 * 384 * 2);
  bf16* wFc1   = (bf16*)alloc(12L * 1536 * 384 * 2);
  bf16* wFc2   = (bf16*)alloc(12L * 384 * 1536 * 2);
  bf16* wCaQ   = (bf16*)alloc(2L * 384 * 384 * 2);
  bf16* wCaK   = (bf16*)alloc(2L * 384 * 384 * 2);
  bf16* wCaV   = (bf16*)alloc(2L * 384 * 384 * 2);
  bf16* wCaPr  = (bf16*)alloc(2L * 384 * 384 * 2);
  bf16* wCaF1  = (bf16*)alloc(2L * 1536 * 384 * 2);
  bf16* wCaF2  = (bf16*)alloc(2L * 384 * 1536 * 2);
  bf16* wHead  = (bf16*)alloc(1000L * 384 * 2);
  bf16*  Ap  = (bf16*)alloc((long)ROWS * PDm * 2);
  float* X   = (float*)alloc((long)ROWS * Dm * 4);
  bf16*  Y   = (bf16*)alloc((long)UROWS * Dm * 2);
  bf16*  QKV = (bf16*)alloc((long)ROWS * 1152 * 2);
  float* A1  = (float*)alloc((long)Bb * Hn * NP * NPAD * 4);
  bf16*  A2  = (bf16*)alloc((long)Bb * Hn * NP * NPAD * 2);
  bf16*  VT  = (bf16*)alloc((long)Bb * Hn * HD * NPAD * 2);
  bf16*  Obf = (bf16*)alloc((long)ROWS * Dm * 2);
  bf16*  Hbf = (bf16*)alloc((long)ROWS * MLPm * 2);
  float* U   = (float*)alloc((long)UROWS * Dm * 4);
  float* CLS = (float*)alloc(32L * Dm * 4);
  float* QC  = (float*)alloc(32L * Dm * 4);
  float* KC  = (float*)alloc((long)UROWS * Dm * 4);
  float* VC  = (float*)alloc((long)UROWS * Dm * 4);
  bf16*  QY  = (bf16*)alloc(32L * Dm * 2);
  bf16*  CAO = (bf16*)alloc(32L * Dm * 2);
  bf16*  CY  = (bf16*)alloc(32L * Dm * 2);
  bf16*  CH  = (bf16*)alloc(32L * MLPm * 2);

  auto cvt = [&](const float* s, bf16* d, long n) {
    k_cvt<<<gsz(n), 256, 0, stream>>>(s, d, n);
  };
  // per-call weight conversion to bf16
  cvt(IN[I_PATCH_W], wPatch, 384L * 768);
  cvt(IN[I_QKVW],  wQkv,  12L * 1152 * 384);
  cvt(IN[I_PROJW], wProj, 12L * 384 * 384);
  cvt(IN[I_FC1W],  wFc1,  12L * 1536 * 384);
  cvt(IN[I_FC2W],  wFc2,  12L * 384 * 1536);
  cvt(IN[I_CQW],   wCaQ,  2L * 384 * 384);
  cvt(IN[I_CKW],   wCaK,  2L * 384 * 384);
  cvt(IN[I_CVW],   wCaV,  2L * 384 * 384);
  cvt(IN[I_CPROJW], wCaPr, 2L * 384 * 384);
  cvt(IN[I_CFC1W], wCaF1, 2L * 1536 * 384);
  cvt(IN[I_CFC2W], wCaF2, 2L * 384 * 1536);
  cvt(IN[I_HEADW], wHead, 1000L * 384);

  // patch embed
  k_patchify<<<gsz((long)ROWS * PDm), 256, 0, stream>>>(IN[I_X], Ap, (long)ROWS * PDm);
  launch_gemm(stream, false, Ap, wPatch, IN[I_PATCH_B], nullptr, nullptr,
              X, nullptr, ROWS, Dm, PDm, PDm, PDm, PDm, PDm, Dm);
  k_addpos<<<gsz((long)ROWS * Dm), 256, 0, stream>>>(X, IN[I_POS], (long)ROWS * Dm);

  const long HSf = (long)NP * NPAD;      // per-head attn slice
  const long VSf = (long)HD * NPAD;      // per-head vT slice
  for (int l = 0; l < 12; ++l) {
    const float* n1w = IN[I_N1W] + l * Dm;   const float* n1b = IN[I_N1B] + l * Dm;
    const float* n2w = IN[I_N2W] + l * Dm;   const float* n2b = IN[I_N2B] + l * Dm;
    const float* plw = IN[I_PLW] + l * 64;   const float* plb = IN[I_PLB] + l * 8;
    const float* pww = IN[I_PWW] + l * 64;   const float* pwb = IN[I_PWB] + l * 8;

    k_layernorm<<<ROWS, 128, 0, stream>>>(X, n1w, n1b, Y, Dm);
    // QKV (no bias) -> bf16
    launch_gemm(stream, false, Y, wQkv + (long)l * 1152 * 384,
                nullptr, nullptr, nullptr, nullptr, QKV,
                ROWS, 1152, Dm, Dm, Dm, Dm, Dm, 1152);
    // scores: per (b,h) 196x196, K=48 (loop padded to 64, zero-filled)
    launch_gemm(stream, false, QKV, QKV + Dm, nullptr, nullptr, nullptr,
                A1, nullptr, NP, NP, 64, HD, HD, 1152, 1152, NPAD,
                Bb * Hn, Hn,
                (long)NP * 1152, HD, (long)NP * 1152, HD, Hn * HSf, HSf);
    k_talking_softmax<<<Bb * NP, 256, 0, stream>>>(A1, A2, plw, plb, pww, pwb);
    // V -> vT[b][h][48][224] (zero-padded), then attn @ V as A @ W^T
    k_transpose_v<<<gsz((long)Bb * Hn * VSf), 256, 0, stream>>>(QKV, VT, (long)Bb * Hn * VSf);
    launch_gemm(stream, false, A2, VT, nullptr, nullptr, nullptr,
                nullptr, Obf, NP, HD, NPAD, NPAD, NPAD, NPAD, NPAD, Dm,
                Bb * Hn, Hn,
                Hn * HSf, HSf, Hn * VSf, VSf, (long)NP * Dm, HD);
    // proj + layer-scale residual into X
    launch_gemm(stream, false, Obf, wProj + (long)l * 384 * 384,
                IN[I_PROJB] + l * Dm, X, IN[I_G1] + l * Dm, X, nullptr,
                ROWS, Dm, Dm, Dm, Dm, Dm, Dm, Dm);
    // MLP
    k_layernorm<<<ROWS, 128, 0, stream>>>(X, n2w, n2b, Y, Dm);
    launch_gemm(stream, true, Y, wFc1 + (long)l * 1536 * 384,
                IN[I_FC1B] + l * MLPm, nullptr, nullptr, nullptr, Hbf,
                ROWS, MLPm, Dm, Dm, Dm, Dm, Dm, MLPm);
    launch_gemm(stream, false, Hbf, wFc2 + (long)l * 384 * 1536,
                IN[I_FC2B] + l * Dm, X, IN[I_G2] + l * Dm, X, nullptr,
                ROWS, Dm, MLPm, MLPm, MLPm, MLPm, MLPm, Dm);
  }

  // class attention
  k_init_cls<<<gsz(32L * Dm), 256, 0, stream>>>(CLS, IN[I_CLS], 32L * Dm);
  for (int i = 0; i < 2; ++i) {
    k_build_u<<<gsz((long)UROWS * Dm), 256, 0, stream>>>(U, CLS, X, (long)UROWS * Dm);
    k_layernorm<<<UROWS, 128, 0, stream>>>(U, IN[I_CN1W] + i * Dm, IN[I_CN1B] + i * Dm, Y, Dm);
    k_gather_rows<<<gsz(32L * Dm), 256, 0, stream>>>(QY, Y, 32L * Dm);
    launch_gemm(stream, false, QY, wCaQ + (long)i * 384 * 384,
                nullptr, nullptr, nullptr, QC, nullptr, 32, Dm, Dm, Dm, Dm, Dm, Dm, Dm);
    launch_gemm(stream, false, Y, wCaK + (long)i * 384 * 384,
                nullptr, nullptr, nullptr, KC, nullptr, UROWS, Dm, Dm, Dm, Dm, Dm, Dm, Dm);
    launch_gemm(stream, false, Y, wCaV + (long)i * 384 * 384,
                nullptr, nullptr, nullptr, VC, nullptr, UROWS, Dm, Dm, Dm, Dm, Dm, Dm, Dm);
    k_ca_attn<<<Bb * Hn, 256, 0, stream>>>(QC, KC, VC, CAO);
    launch_gemm(stream, false, CAO, wCaPr + (long)i * 384 * 384,
                IN[I_CPROJB] + i * Dm, CLS, IN[I_CG1] + i * Dm, CLS, nullptr,
                32, Dm, Dm, Dm, Dm, Dm, Dm, Dm);
    k_layernorm<<<32, 128, 0, stream>>>(CLS, IN[I_CN2W] + i * Dm, IN[I_CN2B] + i * Dm, CY, Dm);
    launch_gemm(stream, true, CY, wCaF1 + (long)i * 1536 * 384,
                IN[I_CFC1B] + i * MLPm, nullptr, nullptr, nullptr, CH,
                32, MLPm, Dm, Dm, Dm, Dm, Dm, MLPm);
    launch_gemm(stream, false, CH, wCaF2 + (long)i * 384 * 1536,
                IN[I_CFC2B] + i * Dm, CLS, IN[I_CG2] + i * Dm, CLS, nullptr,
                32, Dm, MLPm, MLPm, MLPm, MLPm, MLPm, Dm);
  }

  // final norm + head (only cls row matters)
  k_layernorm<<<32, 128, 0, stream>>>(CLS, IN[I_NORMW], IN[I_NORMB], CY, Dm);
  launch_gemm(stream, false, CY, wHead, IN[I_HEADB], nullptr, nullptr,
              (float*)d_out, nullptr, 32, NCLS, Dm, Dm, Dm, Dm, Dm, NCLS);
}
